// Codebook_18038862643696
// MI455X (gfx1250) — matrix operand order repaired and checked by reading, not compiled, not run on hardware
//
#include <hip/hip_runtime.h>
#include <hip/hip_bf16.h>

// ---------------------------------------------------------------------------
// VQ codebook assignment for MI455X (gfx1250, wave32, WMMA + TDM/async copy).
//   N=524288 tokens, D=384, K=512 codes.
//   - GEMM (zn @ Wn^T) in bf16 WMMA (v_wmma_f32_16x16x32_bf16), f32 accum:
//     206 GFLOP vs 1.6 GB HBM -> memory-bound (~69us floor) on the wide-K
//     bf16 matrix path; f32 WMMA (16x16x4) would be 8x the matrix instrs.
//   - Codebook fragments (12 KB per 16-column group) staged ONCE per block
//     into LDS (double-buffered) via the Tensor Data Mover / async-to-LDS
//     path, instead of each of the 8 waves re-streaming them from L2 (8x
//     L2 traffic reduction; keeps the WMMA pipe fed from LDS).
//   - LDS layouts padded/swizzled to avoid bank conflicts.
// ---------------------------------------------------------------------------

#define N_TOK  524288
#define DIM    384
#define ZPITCH 392          // padded LDS row pitch (ushorts) -> breaks 768B-stride conflicts
#define KCB    512
#define BETA   0.25f
#define EPSN   1e-12f

#define ROWS_PER_WAVE   16
#define WAVES_PER_BLOCK 8
#define ROWS_PER_BLOCK  (ROWS_PER_WAVE * WAVES_PER_BLOCK)   // 128
#define NGROUPS         (KCB / 16)                           // 32
#define NKSTEP          (DIM / 32)                           // 12
#define GROUP_BYTES     (NKSTEP * 32 * 32)                   // 12288 B per column group
#define GROUP_U4        (GROUP_BYTES / 16)                   // 768 16-byte chunks

typedef __attribute__((ext_vector_type(16))) __bf16        v16bf;
typedef __attribute__((ext_vector_type(8)))  float         v8f;
typedef __attribute__((ext_vector_type(4)))  unsigned int  u32x4;
typedef __attribute__((ext_vector_type(8)))  int           i32x8;
typedef __attribute__((ext_vector_type(4)))  int           i32x4;

struct alignas(16) U4 { unsigned int x, y, z, w; };
struct U8 { U4 lo; U4 hi; };
union  F16 { U8 u; v16bf v; };

#define AS1 __attribute__((address_space(1)))
#define AS3 __attribute__((address_space(3)))

// Staging path selection (all guarded so the TU always compiles):
//   0 = Tensor Data Mover (one descriptor moves the whole 12 KB group)
//   1 = global_load_async_to_lds_b128 (ASYNCcnt)
//   2 = plain load + ds_store fallback
#if __has_builtin(__builtin_amdgcn_tensor_load_to_lds)
  #define STAGE_MODE 0
#elif __has_builtin(__builtin_amdgcn_global_load_async_to_lds_b128)
  #define STAGE_MODE 1
#else
  #define STAGE_MODE 2
#endif

// round-to-nearest-even f32 -> bf16
static __device__ __forceinline__ unsigned int f2bf(float f) {
    unsigned int u = __float_as_uint(f);
    return (u + 0x7FFFu + ((u >> 16) & 1u)) >> 16;
}

static __device__ __forceinline__ float wave_sum(float v) {
    #pragma unroll
    for (int m = 16; m >= 1; m >>= 1) v += __shfl_xor(v, m, 32);
    return v;
}

// ---------------------------------------------------------------------------
// Stage one 12 KB column group (contiguous in Wfrag) into an LDS buffer.
// ---------------------------------------------------------------------------
static __device__ __forceinline__ void stage_group(const U4* __restrict__ gsrc,
                                                   U4* ldst, int tid)
{
#if STAGE_MODE == 0
    // One TDM descriptor, issued by wave 0: 1D tile of 1536 x 8B elements.
    if (tid < 32) {
        const unsigned long long ga = (unsigned long long)(size_t)gsrc;
        const unsigned int       la = (unsigned int)(size_t)(void*)ldst;
        u32x4 g0;
        g0.x = 1u;                                            // count=1, user mode
        g0.y = la;                                            // lds_addr (bytes)
        g0.z = (unsigned int)ga;                              // global_addr[31:0]
        g0.w = (unsigned int)((ga >> 32) & 0x1FFFFFFu)        // global_addr[56:32]
             | (2u << 30);                                    // type=2 ("image")
        const unsigned int nelem = GROUP_BYTES / 8;           // 1536 8-byte elems
        i32x8 g1;
        g1[0] = (int)(3u << 16);                              // data_size=3 (8B), mask=0
        g1[1] = (int)((nelem & 0xFFFFu) << 16);               // tensor_dim0 lo16
        g1[2] = (int)(1u << 16);                              // tensor_dim0 hi | tensor_dim1=1
        g1[3] = (int)((nelem & 0xFFFFu) << 16);               // tensor_dim1 hi | tile_dim0
        g1[4] = 1;                                            // tile_dim1=1, tile_dim2=0
        g1[5] = (int)nelem;                                   // tensor_dim0_stride lo32
        g1[6] = 0;
        g1[7] = 0;
        i32x4 gz = {0, 0, 0, 0};
  #if __clang_major__ >= 23
        i32x8 gz8 = {};
        __builtin_amdgcn_tensor_load_to_lds(g0, g1, gz, gz, gz8, 0);
  #else
        __builtin_amdgcn_tensor_load_to_lds(g0, g1, gz, gz, 0);
  #endif
    }
#elif STAGE_MODE == 1
    #pragma unroll
    for (int i = 0; i < GROUP_U4 / 256; ++i) {                // 3 x b128 per thread
        const int idx = tid + i * 256;
        __builtin_amdgcn_global_load_async_to_lds_b128(
            (const AS1 void*)(unsigned long long)(size_t)(gsrc + idx),
            (AS3 void*)(unsigned int)(size_t)(void*)(ldst + idx), 0, 0);
    }
#else
    #pragma unroll
    for (int i = 0; i < GROUP_U4 / 256; ++i) {
        const int idx = tid + i * 256;
        ldst[idx] = gsrc[idx];
    }
#endif
}

static __device__ __forceinline__ void stage_wait_and_barrier(int tid)
{
#if STAGE_MODE == 0
    if (tid < 32) __builtin_amdgcn_s_wait_tensorcnt(0);
#elif STAGE_MODE == 1
  #if __has_builtin(__builtin_amdgcn_s_wait_asynccnt)
    __builtin_amdgcn_s_wait_asynccnt(0);
  #else
    asm volatile("s_wait_asynccnt 0x0" ::: "memory");
  #endif
#endif
    __syncthreads();
}

// ---------------------------------------------------------------------------
// Kernel 1: normalize codebook rows -> f32 Wn, ||wn||^2, and a bf16 B-fragment
// buffer. Fragment layout (copied verbatim to LDS by the DMA):
//   group g = 16 columns, 12 ksteps; slot (t,lane) = 32 bytes at
//   chunk indices slot*2 + swz and slot*2 + (1-swz), swz = (lane>>3)&1
//   (XOR swizzle spreads the 32B/lane stride across all LDS bank groups).
//   lane n: column 16g+n, K 32t+0..15 ; lane n+16: same column, K 32t+16..31.
// ---------------------------------------------------------------------------
__global__ __launch_bounds__(32) void k_norm_codebook(
    const float* __restrict__ W,
    float* __restrict__ Wn,
    float* __restrict__ wn2,
    unsigned short* __restrict__ Wfrag)
{
    const int k    = blockIdx.x;
    const int lane = threadIdx.x;
    const float* wp = W + (size_t)k * DIM;

    float vals[NKSTEP];
    float ss = 0.f;
    #pragma unroll
    for (int j = 0; j < NKSTEP; ++j) {
        float v = wp[lane + 32 * j];
        vals[j] = v;
        ss += v * v;
    }
    ss = wave_sum(ss);
    const float scale = 1.f / fmaxf(sqrtf(ss), EPSN);

    #pragma unroll
    for (int j = 0; j < NKSTEP; ++j)
        Wn[(size_t)k * DIM + lane + 32 * j] = vals[j] * scale;
    if (lane == 0) wn2[k] = ss * scale * scale;

    const int g = k >> 4, n = k & 15;
    if (lane < 24) {   // 12 ksteps x 2 half-chunks, one slot per lane
        const int t = lane >> 1, hi = lane & 1;
        const int destLane = n + 16 * hi;
        unsigned int pk[8];
        #pragma unroll
        for (int e = 0; e < 8; ++e) {
            float a = wp[32 * t + 16 * hi + 2 * e]     * scale;
            float b = wp[32 * t + 16 * hi + 2 * e + 1] * scale;
            pk[e] = f2bf(a) | (f2bf(b) << 16);
        }
        U4 q0 = { pk[0], pk[1], pk[2], pk[3] };
        U4 q1 = { pk[4], pk[5], pk[6], pk[7] };
        const int swz = (destLane >> 3) & 1;
        U4* dst = (U4*)Wfrag + ((size_t)g * GROUP_U4)
                             + ((size_t)(t * 32 + destLane)) * 2;
        dst[swz]     = q0;
        dst[1 - swz] = q1;
    }
}

// ---------------------------------------------------------------------------
// Kernel 2: 128-row tile per block. Normalize z rows -> bf16 LDS tile, then
// sweep 32 column groups with WMMA; B fragments double-buffered in LDS via
// TDM/async copy; argmin -> gather Wn[idx], idx, loss partial.
// ---------------------------------------------------------------------------
__global__ __launch_bounds__(256) void k_assign(
    const float* __restrict__ z,
    const float* __restrict__ Wn,
    const float* __restrict__ wn2,
    const unsigned short* __restrict__ Wfrag,
    float* __restrict__ zq_out,
    float* __restrict__ idx_out,
    float* __restrict__ lossPartial)
{
    __shared__ __align__(16) unsigned short zsh[ROWS_PER_BLOCK * ZPITCH]; // ~98 KB
    __shared__ U4   wlds[2][GROUP_U4];                                    // 24 KB
    __shared__ float zn2s[ROWS_PER_BLOCK];
    __shared__ float lred[WAVES_PER_BLOCK];

    const int tid      = threadIdx.x;
    const int lane     = tid & 31;
    const int wave     = tid >> 5;
    const int rowBase  = blockIdx.x * ROWS_PER_BLOCK + wave * ROWS_PER_WAVE;
    const int lrowBase = wave * ROWS_PER_WAVE;

    // --- Normalize 16 rows (coalesced float4), bf16 zn into LDS ------------
    for (int r = 0; r < ROWS_PER_WAVE; ++r) {
        const float* zp = z + (size_t)(rowBase + r) * DIM;
        float4 v0 = *(const float4*)(zp +       lane * 4);
        float4 v1 = *(const float4*)(zp + 128 + lane * 4);
        float4 v2 = *(const float4*)(zp + 256 + lane * 4);
        float ss = v0.x*v0.x + v0.y*v0.y + v0.z*v0.z + v0.w*v0.w
                 + v1.x*v1.x + v1.y*v1.y + v1.z*v1.z + v1.w*v1.w
                 + v2.x*v2.x + v2.y*v2.y + v2.z*v2.z + v2.w*v2.w;
        ss = wave_sum(ss);
        const float scale = 1.f / fmaxf(sqrtf(ss), EPSN);
        if (lane == 0) zn2s[lrowBase + r] = ss * scale * scale;

        unsigned short* dsh = &zsh[(lrowBase + r) * ZPITCH];
        uint2 p0 = make_uint2(f2bf(v0.x*scale) | (f2bf(v0.y*scale) << 16),
                              f2bf(v0.z*scale) | (f2bf(v0.w*scale) << 16));
        uint2 p1 = make_uint2(f2bf(v1.x*scale) | (f2bf(v1.y*scale) << 16),
                              f2bf(v1.z*scale) | (f2bf(v1.w*scale) << 16));
        uint2 p2 = make_uint2(f2bf(v2.x*scale) | (f2bf(v2.y*scale) << 16),
                              f2bf(v2.z*scale) | (f2bf(v2.w*scale) << 16));
        *(uint2*)(dsh +       lane * 4) = p0;
        *(uint2*)(dsh + 128 + lane * 4) = p1;
        *(uint2*)(dsh + 256 + lane * 4) = p2;
    }

    // --- Kick off staging of group 0 while A-fragments are gathered --------
    stage_group((const U4*)Wfrag, &wlds[0][0], tid);

    // --- Hoist all 12 A fragments (16x32 bf16) into VGPRs ------------------
    const int hi = lane >> 4;
    const int nl = lane & 15;
    const int arow = lrowBase + nl;
    F16 Afr[NKSTEP];
    #pragma unroll
    for (int t = 0; t < NKSTEP; ++t) {
        const unsigned short* as = &zsh[arow * ZPITCH + 32 * t + 16 * hi];
        Afr[t].u.lo = *(const U4*)(as);
        Afr[t].u.hi = *(const U4*)(as + 8);
    }

    stage_wait_and_barrier(tid);   // group 0 resident

    // --- WMMA sweep: score[row,col] = ||wn||^2 - 2*dot ----------------------
    float bests[8];
    int   bestc[8];
    #pragma unroll
    for (int i = 0; i < 8; ++i) { bests[i] = 3.4e38f; bestc[i] = 0; }

    const int swz = (lane >> 3) & 1;
    for (int g = 0; g < NGROUPS; ++g) {
        const int cur = g & 1;
        if (g + 1 < NGROUPS)
            stage_group((const U4*)Wfrag + (size_t)(g + 1) * GROUP_U4,
                        &wlds[cur ^ 1][0], tid);

        v8f acc = {};
        const float w2 = wn2[g * 16 + nl];
        const U4* wb = &wlds[cur][0];
        #pragma unroll
        for (int t = 0; t < NKSTEP; ++t) {
            const U4* bs = wb + (t * 32 + lane) * 2;
            F16 B;
            B.u.lo = bs[swz];
            B.u.hi = bs[1 - swz];
            acc = __builtin_amdgcn_wmma_f32_16x16x32_bf16(
                      false, Afr[t].v, false, B.v, (short)0, acc, false, false);
        }
        #pragma unroll
        for (int i = 0; i < 8; ++i) {
            float s = w2 - 2.f * acc[i];            // col = 16g+nl, row = i+8*hi
            if (s < bests[i]) { bests[i] = s; bestc[i] = g * 16 + nl; }
        }

        if (g + 1 < NGROUPS) stage_wait_and_barrier(tid);
    }

    // --- Cross-lane argmin over the 16 columns per lane-half ----------------
    #pragma unroll
    for (int i = 0; i < 8; ++i) {
        float s = bests[i];
        int   c = bestc[i];
        #pragma unroll
        for (int m = 1; m < 16; m <<= 1) {
            float os = __shfl_xor(s, m, 32);
            int   oc = __shfl_xor(c, m, 32);
            if (os < s || (os == s && oc < c)) { s = os; c = oc; }
        }
        bests[i] = s;
        bestc[i] = c;   // uniform across each lane-half
    }

    // --- idx output + loss partial ------------------------------------------
    float lsum = 0.f;
    if (nl == 0) {  // lanes 0 (rows 0..7) and 16 (rows 8..15)
        #pragma unroll
        for (int i = 0; i < 8; ++i) {
            const int r = i + 8 * hi;
            lsum += zn2s[lrowBase + r] + bests[i];  // ||zn||^2 + ||wn||^2 - 2dot
            idx_out[rowBase + r] = (float)bestc[i];
        }
    }
    lsum += __shfl_xor(lsum, 16, 32);
    if (lane == 0) lred[wave] = lsum;

    // --- Gather z_q = Wn[idx] (STE forward value) ---------------------------
    #pragma unroll
    for (int r = 0; r < ROWS_PER_WAVE; ++r) {
        const int i = r & 7;
        const int c = __shfl(bestc[i], (r < 8) ? 0 : 16, 32);
        const float* wrow = Wn + (size_t)c * DIM;
        float* orow = zq_out + (size_t)(rowBase + r) * DIM;
        *(float4*)(orow +       lane * 4) = *(const float4*)(wrow +       lane * 4);
        *(float4*)(orow + 128 + lane * 4) = *(const float4*)(wrow + 128 + lane * 4);
        *(float4*)(orow + 256 + lane * 4) = *(const float4*)(wrow + 256 + lane * 4);
    }

    __syncthreads();
    if (tid == 0) {
        float t = 0.f;
        #pragma unroll
        for (int w = 0; w < WAVES_PER_BLOCK; ++w) t += lred[w];
        lossPartial[blockIdx.x] = t;
    }
}

// ---------------------------------------------------------------------------
// Kernel 3: deterministic reduction of per-block loss partials.
// ---------------------------------------------------------------------------
__global__ __launch_bounds__(256) void k_loss(
    const float* __restrict__ partial, int nPartial, float* __restrict__ lossOut)
{
    __shared__ float sh[256];
    float s = 0.f;
    for (int i = threadIdx.x; i < nPartial; i += 256) s += partial[i];
    sh[threadIdx.x] = s;
    __syncthreads();
    for (int step = 128; step > 0; step >>= 1) {
        if (threadIdx.x < step) sh[threadIdx.x] += sh[threadIdx.x + step];
        __syncthreads();
    }
    if (threadIdx.x == 0)
        *lossOut = (1.f + BETA) * sh[0] / ((float)N_TOK * (float)DIM);
}

// ---------------------------------------------------------------------------
extern "C" void kernel_launch(void* const* d_in, const int* in_sizes, int n_in,
                              void* d_out, int out_size, void* d_ws, size_t ws_size,
                              hipStream_t stream)
{
    const float* z = (const float*)d_in[0];   // [N, D]
    const float* W = (const float*)d_in[1];   // [K, D]

    float* out   = (float*)d_out;
    float* zq    = out;
    float* idxo  = out + (size_t)N_TOK * DIM;
    float* losso = idxo + N_TOK;

    char* ws = (char*)d_ws;
    float*          Wn    = (float*)ws;                     // 786432 B
    float*          wn2   = (float*)(ws + 786432);          //   2048 B
    unsigned short* Wfrag = (unsigned short*)(ws + 788480); // 393216 B
    float*          lp    = (float*)(ws + 788480 + 393216); //  16384 B

    const int nBlocks = N_TOK / ROWS_PER_BLOCK;   // 4096

    k_norm_codebook<<<KCB, 32, 0, stream>>>(W, Wn, wn2, Wfrag);
    k_assign<<<nBlocks, 256, 0, stream>>>(z, Wn, wn2, Wfrag, zq, idxo, lp);
    k_loss<<<1, 256, 0, stream>>>(lp, nBlocks, losso);
}
